// FBP_13511967113530
// MI455X (gfx1250) — matrix operand order repaired and checked by reading, not compiled, not run on hardware
//
#include <hip/hip_runtime.h>
#include <hip/hip_bf16.h>
#include <math.h>

#define CHAN  512
#define VIEWS 360
#define BATCH 8
#define PADN  100
#define ILEN  1024
#define NCOL  (BATCH * VIEWS)   // 2880 sinogram columns

typedef __attribute__((ext_vector_type(16))) _Float16 v16h;
typedef __attribute__((ext_vector_type(8)))  _Float16 v8h;
typedef __attribute__((ext_vector_type(8)))  float    v8f;

// ---------------------------------------------------------------------------
// h[n] = real(ifft(|fftfreq(1024)|))[n] = (1/N) * sum_k ramp[k] cos(2*pi*k*n/N)
// Integer (k*n) mod 1024 keeps the cos argument exactly reduced.
// ---------------------------------------------------------------------------
__global__ void compute_h_kernel(float* __restrict__ h) {
    int n = blockIdx.x * blockDim.x + threadIdx.x;
    if (n >= ILEN) return;
    const float w = 6.28318530717958647692f / (float)ILEN;
    float s = 0.0f;
    for (int k = 0; k < ILEN; ++k) {
        int kr = (k < ILEN - k) ? k : (ILEN - k);          // min(k, N-k)
        float ramp = (float)kr * (1.0f / (float)ILEN);     // |fftfreq|
        int arg = (k * n) & (ILEN - 1);                    // exact mod 1024
        s += ramp * cosf(w * (float)arg);
    }
    h[n] = s * (1.0f / (float)ILEN);
}

// ---------------------------------------------------------------------------
// G[n,m] = h[(n-m) mod 1024]
//        + (m==0)   : sum_j (j/100)      * h[(n+100-j) mod 1024]   (head taper)
//        + (m==511) : sum_j ((99-j)/100) * h[(n-512-j) mod 1024]   (tail taper)
// Stored as split f16: G = Ghi + Glo (Glo = residual), for WMMA with ~fp32 acc.
// ---------------------------------------------------------------------------
__global__ void compute_G_kernel(const float* __restrict__ h,
                                 _Float16* __restrict__ Ghi,
                                 _Float16* __restrict__ Glo) {
    int t = blockIdx.x * blockDim.x + threadIdx.x;
    if (t >= CHAN * CHAN) return;
    int n = t >> 9;
    int m = t & (CHAN - 1);
    float g = h[(n - m) & (ILEN - 1)];
    if (m == 0) {
        for (int j = 0; j < PADN; ++j)
            g += (float)j * (1.0f / PADN) * h[(n + PADN - j) & (ILEN - 1)];
    }
    if (m == CHAN - 1) {
        for (int j = 0; j < PADN; ++j)
            g += (float)(PADN - 1 - j) * (1.0f / PADN) * h[(n - CHAN - j) & (ILEN - 1)];
    }
    _Float16 hi = (_Float16)g;
    Ghi[t] = hi;
    Glo[t] = (_Float16)(g - (float)hi);
}

// ---------------------------------------------------------------------------
// Pack x[B,1,512,360] -> X^T[(b*360+v)][k] split f16 (K contiguous so B-matrix
// WMMA fragments are b128 loads).
// ---------------------------------------------------------------------------
__global__ void pack_x_kernel(const float* __restrict__ x,
                              _Float16* __restrict__ XTh,
                              _Float16* __restrict__ XTl) {
    int t = blockIdx.x * blockDim.x + threadIdx.x;   // NCOL*CHAN threads
    int col = t >> 9;                                // (b*360 + v)
    int k   = t & (CHAN - 1);
    int b = col / VIEWS;
    int v = col - b * VIEWS;
    float val = x[(size_t)b * CHAN * VIEWS + (size_t)k * VIEWS + v];
    _Float16 hi = (_Float16)val;
    XTh[t] = hi;
    XTl[t] = (_Float16)(val - (float)hi);
}

// ---------------------------------------------------------------------------
// Filter GEMM: D(512x2880) = G(512x512) * X(512x2880), split-f16 3-product.
// One wave computes a 16(M) x 64(N) strip = four 16x16 D tiles sharing one A
// fragment (4x less A traffic, 12 WMMAs per K-step per wave). K runs in 16
// steps of 32. Output stored transposed Dt[(b,v)][y] (y contiguous) for the
// backprojection pass. Fragment layouts follow CDNA5 ISA 7.12.2 (wave32).
// ---------------------------------------------------------------------------
__global__ __launch_bounds__(256) void filter_gemm_kernel(
        const _Float16* __restrict__ Ghi, const _Float16* __restrict__ Glo,
        const _Float16* __restrict__ XTh, const _Float16* __restrict__ XTl,
        float* __restrict__ Dt) {
    int wave = threadIdx.x >> 5;
    int lane = threadIdx.x & 31;
    int tile = blockIdx.x * 8 + wave;     // 32 * 45 = 1440 wave-strips
    int mt  = tile & 31;                  // M tile (channel rows of G)
    int ng  = tile >> 5;                  // N group of 4 tiles (0..44)
    int nt0 = ng * 4;
    int half = lane >> 4;                 // lane half selects K sub-range
    int lidx = lane & 15;                 // row (A) / col (B,D) within tile

    const _Float16* arow_h  = Ghi + ((size_t)(mt * 16 + lidx) << 9);
    const _Float16* arow_l  = Glo + ((size_t)(mt * 16 + lidx) << 9);
    const _Float16* bbase_h = XTh + ((size_t)(nt0 * 16 + lidx) << 9);
    const _Float16* bbase_l = XTl + ((size_t)(nt0 * 16 + lidx) << 9);

    v8f c[4] = {};
    for (int ks = 0; ks < 16; ++ks) {
        int ko = ks << 5;
        // A fragment: lane holds row m, K = ko + 8*half + [0..7] and +16
        v8h a0h = *(const v8h*)(arow_h + ko + 8 * half);
        v8h a1h = *(const v8h*)(arow_h + ko + 16 + 8 * half);
        v8h a0l = *(const v8h*)(arow_l + ko + 8 * half);
        v8h a1l = *(const v8h*)(arow_l + ko + 16 + 8 * half);
        v16h aH, aL;
#pragma unroll
        for (int i = 0; i < 8; ++i) {
            aH[i] = a0h[i]; aH[i + 8] = a1h[i];
            aL[i] = a0l[i]; aL[i + 8] = a1l[i];
        }
#pragma unroll
        for (int j = 0; j < 4; ++j) {
            // B fragment for N tile nt0+j: lane holds col n, K = ko + 8*half + [0..7] and +16
            const _Float16* bh = bbase_h + (size_t)j * (16 * CHAN);
            const _Float16* bl = bbase_l + (size_t)j * (16 * CHAN);
            v8h b0h = *(const v8h*)(bh + ko + 8 * half);
            v8h b1h = *(const v8h*)(bh + ko + 16 + 8 * half);
            v8h b0l = *(const v8h*)(bl + ko + 8 * half);
            v8h b1l = *(const v8h*)(bl + ko + 16 + 8 * half);
            v16h bH, bL;
#pragma unroll
            for (int i = 0; i < 8; ++i) {
                bH[i] = b0h[i]; bH[i + 8] = b1h[i];
                bL[i] = b0l[i]; bL[i + 8] = b1l[i];
            }
            // D += Gh*Xh + Gh*Xl + Gl*Xh   (split-f16, fp32 accumulate)
            c[j] = __builtin_amdgcn_wmma_f32_16x16x32_f16(false, aH, false, bH,
                                                          (short)0, c[j], false, false);
            c[j] = __builtin_amdgcn_wmma_f32_16x16x32_f16(false, aH, false, bL,
                                                          (short)0, c[j], false, false);
            c[j] = __builtin_amdgcn_wmma_f32_16x16x32_f16(false, aL, false, bH,
                                                          (short)0, c[j], false, false);
        }
    }
    // D element v in lane: row = mt*16 + 8*half + v, col = (nt0+j)*16 + lidx.
    // Store transposed: Dt[col][row] -> 8 consecutive floats per lane.
#pragma unroll
    for (int j = 0; j < 4; ++j) {
        float* dp = Dt + ((size_t)((nt0 + j) * 16 + lidx) << 9) + mt * 16 + 8 * half;
#pragma unroll
        for (int v = 0; v < 8; ++v) dp[v] = c[j][v];
    }
}

// ---------------------------------------------------------------------------
// Backprojection: each block = one 16x16 pixel tile of one batch image.
// The 512-float filtered column for each view is double-buffered in LDS via
// gfx1250 async memory->LDS copies (ASYNCcnt): issue view v+1's copy, overlap
// it with view v's pixel updates, sync with s_wait_asynccnt + barrier.
// LDS byte address = low 32 bits of the generic shared-memory pointer
// (ISA 10.2 aperture rule: flat LDS addresses carry the LDS offset in [31:0]).
// ---------------------------------------------------------------------------
__global__ __launch_bounds__(256) void backproject_kernel(
        const float* __restrict__ Dt, float* __restrict__ out) {
    int b = blockIdx.x >> 10;           // batch
    int t = blockIdx.x & 1023;          // 32x32 tiles of 16x16
    int R = ((t >> 5) << 4) + (threadIdx.x >> 4);
    int C = ((t & 31) << 4) + (threadIdx.x & 15);

    __shared__ float pv[2][CHAN];       // double-buffered filtered column

    const float* dbase = Dt + (size_t)b * VIEWS * CHAN;
    unsigned lds0 = (unsigned)(uintptr_t)(void*)&pv[0][0];
    unsigned tid8 = (unsigned)threadIdx.x * 8u;   // 256 lanes x b64 = 2 KB

    // Prologue: async-copy view 0's column into buffer 0.
    asm volatile("global_load_async_to_lds_b64 %0, %1, %2 offset:0"
                 :: "v"(lds0 + tid8), "v"(tid8), "s"(dbase) : "memory");

    float gx = (2.0f * C + 1.0f) * (1.0f / CHAN) - 1.0f;
    float gy = (2.0f * R + 1.0f) * (1.0f / CHAN) - 1.0f;
    float acc = 0.0f;

    for (int v = 0; v < VIEWS; ++v) {
        // Own async copies done, then barrier so every wave's copy is visible.
        asm volatile("s_wait_asynccnt 0x0" ::: "memory");
        __syncthreads();
        if (v + 1 < VIEWS) {            // start next view into the other buffer
            unsigned buf = (unsigned)(v + 1) & 1u;
            asm volatile("global_load_async_to_lds_b64 %0, %1, %2 offset:0"
                         :: "v"(lds0 + buf * 2048u + tid8),
                            "v"((unsigned)(v + 1) * 2048u + tid8),
                            "s"(dbase) : "memory");
        }
        const float* col = pv[v & 1];

        float ang = -1.57079632679489662f +
                    3.14159265358979324f * (float)(v + 1) * (1.0f / VIEWS);
        float sa, ca;
        sincosf(ang, &sa, &ca);
        float sx = ca * gx - sa * gy;
        float sy = sa * gx + ca * gy;
        float ix = sx * 256.0f + 255.5f;   // ((sx+1)*512 - 1)/2
        float iy = sy * 256.0f + 255.5f;
        float x0f = floorf(ix), y0f = floorf(iy);
        float fx = ix - x0f,    fy = iy - y0f;
        int x0 = (int)x0f, y0 = (int)y0f;

        float acc_v = 0.0f;
#pragma unroll
        for (int dy = 0; dy < 2; ++dy) {
            int yc = y0 + dy;
            if (yc >= 0 && yc < CHAN) {
                float wy  = dy ? fy : (1.0f - fy);
                float py  = (float)yc - 255.5f;
                float val = col[yc];
#pragma unroll
                for (int dx = 0; dx < 2; ++dx) {
                    int xc = x0 + dx;
                    if (xc >= 0 && xc < CHAN) {
                        float wx  = dx ? fx : (1.0f - fx);
                        float px  = (float)xc - 255.5f;
                        float fov = (px * px + py * py < 65536.0f) ? 1.0f : 0.0f;
                        acc_v += wy * wx * fov * val;
                    }
                }
            }
        }
        acc += acc_v;
    }
    out[(size_t)b * CHAN * CHAN + (size_t)R * CHAN + C] =
        acc * (3.14159265358979324f / VIEWS);
}

// ---------------------------------------------------------------------------
extern "C" void kernel_launch(void* const* d_in, const int* in_sizes, int n_in,
                              void* d_out, int out_size, void* d_ws, size_t ws_size,
                              hipStream_t stream) {
    (void)in_sizes; (void)n_in; (void)out_size; (void)ws_size;

    const float* x = (const float*)d_in[0];
    float* out = (float*)d_out;
    char* ws = (char*)d_ws;

    // Workspace carve-up (all 16B aligned; ~12.3 MB total, L2 resident):
    const size_t SZ_H   = 4096;                          // 1024 f32 (padded)
    const size_t SZ_G   = (size_t)CHAN * CHAN * 2;       // 512 KB f16
    const size_t SZ_XT  = (size_t)NCOL * CHAN * 2;       // ~2.95 MB f16

    float*    h   = (float*)ws;
    _Float16* Ghi = (_Float16*)(ws + SZ_H);
    _Float16* Glo = (_Float16*)(ws + SZ_H + SZ_G);
    _Float16* XTh = (_Float16*)(ws + SZ_H + 2 * SZ_G);
    _Float16* XTl = (_Float16*)(ws + SZ_H + 2 * SZ_G + SZ_XT);
    float*    Dt  = (float*)   (ws + SZ_H + 2 * SZ_G + 2 * SZ_XT);

    compute_h_kernel  <<<ILEN / 256,          256, 0, stream>>>(h);
    compute_G_kernel  <<<(CHAN * CHAN) / 256, 256, 0, stream>>>(h, Ghi, Glo);
    pack_x_kernel     <<<(NCOL * CHAN) / 256, 256, 0, stream>>>(x, XTh, XTl);
    filter_gemm_kernel<<<(32 * 45) / 8,       256, 0, stream>>>(Ghi, Glo,
                                                                XTh, XTl, Dt);
    backproject_kernel<<<BATCH * 1024,        256, 0, stream>>>(Dt, out);
}
